// CFMB_85985245266261
// MI455X (gfx1250) — matrix operand order repaired and checked by reading, not compiled, not run on hardware
//
#include <hip/hip_runtime.h>
#include <math.h>

// ---------------- problem constants ----------------
#define BB    4
#define CCH   128
#define HGT   64
#define WID   64
#define LLEN  4096          // HGT*WID
#define BL    16384         // BB*LLEN
#define NHEAD 8
#define HID   340
#define DSTATE 16
#define DINNER 256
#define DTRANK 8
#define KDIR  4
#define MTILES 1024         // BL/16

typedef __bf16 bf16_t;
typedef bf16_t v16bf __attribute__((ext_vector_type(16)));
typedef float  v8f   __attribute__((ext_vector_type(8)));

// ---------------- device helpers ----------------
__device__ __forceinline__ float dev_silu(float x) { return x / (1.0f + __expf(-x)); }
__device__ __forceinline__ float dev_gelu(float x) { return 0.5f * x * (1.0f + erff(x * 0.70710678118654752f)); }
__device__ __forceinline__ int scan_spatial(int k, int l) {
    // map scan-order index l (direction k) -> spatial index h*W+w
    switch (k) {
        case 0: return l;
        case 1: return ((l & 63) << 6) | (l >> 6);          // wh order
        case 2: return 4095 - l;                            // reversed hw
        default: { int lp = 4095 - l; return ((lp & 63) << 6) | (lp >> 6); }
    }
}

// ---------------- weight prepack: f32 [N,K] -> bf16 Wp[Npad][Kpad] (zero padded) ----------------
__global__ void k_pack(const float* __restrict__ Wt, bf16_t* __restrict__ Wp,
                       int N, int K, int Npad, int Kpad)
{
    int i = blockIdx.x * blockDim.x + threadIdx.x;
    int total = Npad * Kpad;
    if (i >= total) return;
    int k = i % Kpad, n = i / Kpad;
    Wp[i] = (n < N && k < K) ? (bf16_t)Wt[(size_t)n * K + k] : (bf16_t)0.0f;
}

// ---------------- WMMA GEMM: Y[m, coY+n] = X[row(m), coX+k] * W[n,k]^T + bias[n] ----------------
// M fixed = BL. One wave per 16x16 tile. Unconditional b128 loads, bf16 convert, f32 accumulate.
// Wp is prepacked/zero-padded so the hot loop has no bounds checks.
// dir >= 0: rows gathered through SS2D scan-order permutation (per batch of 4096 rows).
__global__ void gemm_wmma(const float* __restrict__ X, int ldX, int coX,
                          const bf16_t* __restrict__ Wp,   // [Npad][Kpad] bf16
                          const float* __restrict__ bias,  // may be null
                          float* __restrict__ Y, int ldY, int coY,
                          int Ncols, int Npad, int Kpad, int dir)
{
    int wave = (blockIdx.x * blockDim.x + threadIdx.x) >> 5;
    int lane = threadIdx.x & 31;
    int ntiles = Npad >> 4;
    int total  = MTILES * ntiles;
    if (wave >= total) return;
    int mt = wave % MTILES;
    int nt = wave / MTILES;

    // A row for this lane (lane&15 = M row; lane>>4 selects K half k0+16h..+15, same order as B)
    int mrow = mt * 16 + (lane & 15);
    int srow;
    if (dir < 0) srow = mrow;
    else {
        int b = mrow >> 12;
        int l = mrow & 4095;
        srow = (b << 12) + scan_spatial(dir, l);
    }
    const float* xr = X + (size_t)srow * ldX + coX;
    int kb = (lane >> 4) * 16;

    // B: lane&15 indexes N, lane>>4 selects K half (k0+16h..+15), 16 contiguous bf16 = one 32B load
    const bf16_t* wr = Wp + (size_t)(nt * 16 + (lane & 15)) * Kpad + (lane >> 4) * 16;

    v8f acc = {};
    #pragma unroll 2
    for (int k0 = 0; k0 < Kpad; k0 += 32) {
        const float4 a0 = *(const float4*)(xr + k0 + kb);
        const float4 a1 = *(const float4*)(xr + k0 + kb + 4);
        const float4 a2 = *(const float4*)(xr + k0 + kb + 8);
        const float4 a3 = *(const float4*)(xr + k0 + kb + 12);
        v16bf a;
        a[0]  = (bf16_t)a0.x; a[1]  = (bf16_t)a0.y; a[2]  = (bf16_t)a0.z; a[3]  = (bf16_t)a0.w;
        a[4]  = (bf16_t)a1.x; a[5]  = (bf16_t)a1.y; a[6]  = (bf16_t)a1.z; a[7]  = (bf16_t)a1.w;
        a[8]  = (bf16_t)a2.x; a[9]  = (bf16_t)a2.y; a[10] = (bf16_t)a2.z; a[11] = (bf16_t)a2.w;
        a[12] = (bf16_t)a3.x; a[13] = (bf16_t)a3.y; a[14] = (bf16_t)a3.z; a[15] = (bf16_t)a3.w;
        const v16bf bfrag = *(const v16bf*)(wr + k0);
        acc = __builtin_amdgcn_wmma_f32_16x16x32_bf16(false, a, false, bfrag,
                                                      (short)0, acc, false, false);
    }

    // store: VGPR r, lanes 0-15 -> M=r, lanes 16-31 -> M=r+8; N = lane&15
    int nn = nt * 16 + (lane & 15);
    if (nn < Ncols) {
        float bb = bias ? bias[nn] : 0.0f;
        int rbase = mt * 16 + ((lane >> 4) << 3);
        #pragma unroll
        for (int r = 0; r < 8; ++r)
            Y[(size_t)(rbase + r) * ldY + coY + nn] = acc[r] + bb;
    }
}

// ---------------- LayerNorm over last dim (one wave per row) ----------------
__global__ void ln_lastdim(const float* __restrict__ X, const float* __restrict__ g,
                           const float* __restrict__ b, float* __restrict__ Y,
                           int Cn, int rows)
{
    int wid = (blockIdx.x * blockDim.x + threadIdx.x) >> 5;
    int lane = threadIdx.x & 31;
    if (wid >= rows) return;
    const float* xr = X + (size_t)wid * Cn;
    float s = 0.f, s2 = 0.f;
    for (int c = lane; c < Cn; c += 32) { float v = xr[c]; s += v; s2 += v * v; }
    #pragma unroll
    for (int off = 16; off > 0; off >>= 1) { s += __shfl_xor(s, off, 32); s2 += __shfl_xor(s2, off, 32); }
    float mu  = s / (float)Cn;
    float var = s2 / (float)Cn - mu * mu;
    float inv = rsqrtf(var + 1e-5f);
    float* yr = Y + (size_t)wid * Cn;
    for (int c = lane; c < Cn; c += 32) yr[c] = (xr[c] - mu) * inv * g[c] + b[c];
}

// ---------------- depthwise 3x3 conv (channels-last), optional bias/silu ----------------
__global__ void dwconv3(const float* __restrict__ X, int ldX, int coX,
                        const float* __restrict__ Wd, const float* __restrict__ bias,
                        float* __restrict__ Y, int Cn, int act)
{
    long long idx = (long long)blockIdx.x * blockDim.x + threadIdx.x;
    long long total = (long long)BL * Cn;
    if (idx >= total) return;
    int c = (int)(idx % Cn);
    int m = (int)(idx / Cn);
    int l = m & 4095, b = m >> 12;
    int h = l >> 6, w = l & 63;
    float acc = bias ? bias[c] : 0.0f;
    const float* wp = Wd + (size_t)c * 9;
    #pragma unroll
    for (int dy = 0; dy < 3; ++dy) {
        int hh = h + dy - 1;
        if (hh < 0 || hh >= HGT) continue;
        #pragma unroll
        for (int dx = 0; dx < 3; ++dx) {
            int ww = w + dx - 1;
            if (ww < 0 || ww >= WID) continue;
            acc += X[(size_t)((b << 12) + (hh << 6) + ww) * ldX + coX + c] * wp[dy * 3 + dx];
        }
    }
    if (act == 1) acc = dev_silu(acc);
    Y[(size_t)m * Cn + c] = acc;
}

// ---------------- full 3x3 conv (channels-last), bias, optional gelu ----------------
__global__ void conv3x3(const float* __restrict__ X, int Cin, const float* __restrict__ Wt,
                        const float* __restrict__ bias, float* __restrict__ Y, int Cout, int act)
{
    long long idx = (long long)blockIdx.x * blockDim.x + threadIdx.x;
    long long total = (long long)BL * Cout;
    if (idx >= total) return;
    int co = (int)(idx % Cout);
    int m  = (int)(idx / Cout);
    int l = m & 4095, b = m >> 12;
    int h = l >> 6, w = l & 63;
    float acc = bias[co];
    for (int dy = 0; dy < 3; ++dy) {
        int hh = h + dy - 1;
        if (hh < 0 || hh >= HGT) continue;
        for (int dx = 0; dx < 3; ++dx) {
            int ww = w + dx - 1;
            if (ww < 0 || ww >= WID) continue;
            const float* xp = X + (size_t)((b << 12) + (hh << 6) + ww) * Cin;
            const float* wp = Wt + (size_t)co * Cin * 9 + dy * 3 + dx;
            for (int ci = 0; ci < Cin; ++ci) acc += xp[ci] * wp[(size_t)ci * 9];
        }
    }
    if (act == 1) acc = dev_gelu(acc);
    Y[(size_t)m * Cout + co] = acc;
}

// ---------------- channel attention half (one block per (b,head)) ----------------
__global__ void attn_half(const float* __restrict__ Q, const float* __restrict__ KV,
                          const float* __restrict__ temp, float* __restrict__ O)
{
    int b  = blockIdx.x >> 3;
    int hd = blockIdx.x & 7;
    int t  = threadIdx.x;
    __shared__ float sQ[32][17], sK[32][17], sP[16][17], sS[16][17], sNq[16], sNk[16];
    int c  = t >> 4, dd = t & 15;
    float acc = 0.f, nacc = 0.f;
    const size_t qbase  = (size_t)b * LLEN * 128 + hd * 16;
    const size_t kvbase = (size_t)b * LLEN * 256 + hd * 16;

    for (int l0 = 0; l0 < LLEN; l0 += 32) {
        #pragma unroll
        for (int i = 0; i < 2; ++i) {
            int idx = t + i * 256;
            int li = idx >> 4, ci = idx & 15;
            sQ[li][ci] = Q[qbase + (size_t)(l0 + li) * 128 + ci];
            sK[li][ci] = KV[kvbase + (size_t)(l0 + li) * 256 + ci];
        }
        __syncthreads();
        #pragma unroll 8
        for (int li = 0; li < 32; ++li) acc += sQ[li][c] * sK[li][dd];
        if (t < 16) {
            #pragma unroll 8
            for (int li = 0; li < 32; ++li) nacc += sQ[li][t] * sQ[li][t];
        } else if (t < 32) {
            int cc = t - 16;
            #pragma unroll 8
            for (int li = 0; li < 32; ++li) nacc += sK[li][cc] * sK[li][cc];
        }
        __syncthreads();
    }
    sS[c][dd] = acc;
    if (t < 16)       sNq[t]      = fmaxf(sqrtf(nacc), 1e-12f);
    else if (t < 32)  sNk[t - 16] = fmaxf(sqrtf(nacc), 1e-12f);
    __syncthreads();
    if (t < 16) {
        float tp = temp[hd];
        float row[16], mx = -1e30f;
        #pragma unroll
        for (int d = 0; d < 16; ++d) { row[d] = sS[t][d] / (sNq[t] * sNk[d]) * tp; mx = fmaxf(mx, row[d]); }
        float sum = 0.f;
        #pragma unroll
        for (int d = 0; d < 16; ++d) { row[d] = __expf(row[d] - mx); sum += row[d]; }
        #pragma unroll
        for (int d = 0; d < 16; ++d) sP[t][d] = row[d] / sum;
    }
    __syncthreads();
    // O[c,l] = sum_d P[c,d] * V[d,l]
    int c2 = t & 15, li2 = t >> 4;
    for (int l0 = 0; l0 < LLEN; l0 += 32) {
        #pragma unroll
        for (int i = 0; i < 2; ++i) {
            int idx = t + i * 256;
            int li = idx >> 4, ci = idx & 15;
            sK[li][ci] = KV[kvbase + (size_t)(l0 + li) * 256 + 128 + ci];   // V half
        }
        __syncthreads();
        float o0 = 0.f, o1 = 0.f;
        #pragma unroll
        for (int d = 0; d < 16; ++d) { o0 += sP[c2][d] * sK[li2][d]; o1 += sP[c2][d] * sK[li2 + 16][d]; }
        O[(size_t)b * LLEN * 128 + (size_t)(l0 + li2) * 128 + hd * 16 + c2] = o0;
        O[(size_t)b * LLEN * 128 + (size_t)(l0 + li2 + 16) * 128 + hd * 16 + c2] = o1;
        __syncthreads();
    }
}

// ---------------- selective scan: one block per (b,k), 256 lanes = inner channels ----------------
__global__ void sscan(const float* __restrict__ xc,     // [B*L,256] spatial order
                      const float* __restrict__ xdbl,   // [K][B*L][40] scan order
                      const float* __restrict__ delta,  // [K][B*L][256] scan order
                      const float* __restrict__ A_logs, // [K*256,16]
                      const float* __restrict__ Ds,     // [K*256]
                      float* __restrict__ ycomb)        // [B*L,256] spatial, pre-zeroed
{
    int b = blockIdx.x >> 2, k = blockIdx.x & 3;
    int d = threadIdx.x;
    float A[16], h[16];
    #pragma unroll
    for (int n = 0; n < 16; ++n) { A[n] = -__expf(A_logs[(size_t)(k * 256 + d) * 16 + n]); h[n] = 0.f; }
    float Dd = Ds[k * 256 + d];
    __shared__ float sBC[32];
    const size_t dbase = ((size_t)k * BL + (size_t)b * LLEN) * 256;
    const size_t xbase = ((size_t)k * BL + (size_t)b * LLEN) * 40;
    for (int l = 0; l < LLEN; ++l) {
        if (d < 32) sBC[d] = xdbl[xbase + (size_t)l * 40 + 8 + d];
        __syncthreads();
        float dt = delta[dbase + (size_t)l * 256 + d];
        int s = scan_spatial(k, l);
        float u  = xc[((size_t)(b << 12) + s) * 256 + d];
        float du = dt * u;
        float y  = 0.f;
        #pragma unroll
        for (int n = 0; n < 16; ++n) {
            h[n] = h[n] * __expf(dt * A[n]) + du * sBC[n];
            y += h[n] * sBC[16 + n];
        }
        atomicAdd(&ycomb[((size_t)(b << 12) + s) * 256 + d], y + u * Dd);
        __syncthreads();
    }
}

// ---------------- small elementwise kernels ----------------
__global__ void k_zero(float* Y, long long n) {
    long long i = (long long)blockIdx.x * blockDim.x + threadIdx.x;
    if (i < n) Y[i] = 0.0f;
}
__global__ void k_add(const float* A, const float* Bp, float* Y, long long n) {
    long long i = (long long)blockIdx.x * blockDim.x + threadIdx.x;
    if (i < n) Y[i] = A[i] + Bp[i];
}
__global__ void k_scaleadd(const float* A, const float* s, const float* Bp, float* Y, long long n) {
    long long i = (long long)blockIdx.x * blockDim.x + threadIdx.x;
    if (i < n) Y[i] = A[i] * s[0] + Bp[i];
}
__global__ void k_gate(const float* T2, float* G) {   // gelu(x1)*x2, x1|x2 split of 680
    long long i = (long long)blockIdx.x * blockDim.x + threadIdx.x;
    long long n = (long long)BL * HID;
    if (i >= n) return;
    int c = (int)(i % HID); int m = (int)(i / HID);
    float a = T2[(size_t)m * (2 * HID) + c];
    float g = T2[(size_t)m * (2 * HID) + HID + c];
    G[i] = dev_gelu(a) * g;
}
__global__ void k_mulsilu(float* Y, const float* XZ) { // Y *= silu(z), z = XZ[:,256:512]
    long long i = (long long)blockIdx.x * blockDim.x + threadIdx.x;
    long long n = (long long)BL * DINNER;
    if (i >= n) return;
    int c = (int)(i & 255); int m = (int)(i >> 8);
    Y[i] = Y[i] * dev_silu(XZ[(size_t)m * 512 + 256 + c]);
}
__global__ void k_softplus(float* D, const float* dtb) {
    long long i = (long long)blockIdx.x * blockDim.x + threadIdx.x;
    long long n = (long long)KDIR * BL * DINNER;
    if (i >= n) return;
    int k = (int)(i / ((long long)BL * DINNER));
    int d = (int)(i & 255);
    float v = D[i] + dtb[k * DINNER + d];
    D[i] = (v > 20.0f) ? v : log1pf(__expf(v));
}
__global__ void k_nchw2nlc(const float* X, float* Y) {
    long long i = (long long)blockIdx.x * blockDim.x + threadIdx.x;
    long long n = (long long)BB * CCH * LLEN;
    if (i >= n) return;
    int l = (int)(i % LLEN);
    int c = (int)((i / LLEN) % CCH);
    int b = (int)(i / ((long long)CCH * LLEN));
    Y[((size_t)(b << 12) + l) * CCH + c] = X[i];
}
__global__ void k_final(const float* y3, const float* cab, const float* s2, float* out) {
    long long i = (long long)blockIdx.x * blockDim.x + threadIdx.x;
    long long n = (long long)BB * CCH * LLEN;
    if (i >= n) return;
    int l = (int)(i % LLEN);
    int c = (int)((i / LLEN) % CCH);
    int b = (int)(i / ((long long)CCH * LLEN));
    size_t mi = ((size_t)(b << 12) + l) * CCH + c;
    out[i] = y3[mi] * s2[0] + cab[mi];
}

// ---------------- host-side helpers ----------------
static inline int eblocks(long long n) { return (int)((n + 255) / 256); }

struct PW { bf16_t* p; int Npad; int Kpad; };

static void launch_gemm(hipStream_t st, const float* X, int ldX, int coX,
                        PW w, const float* bias,
                        float* Y, int ldY, int coY, int Ncols, int dir)
{
    int ntiles = w.Npad >> 4;
    int waves  = MTILES * ntiles;
    int blocks = (waves + 7) / 8;           // 8 waves (256 threads) per block
    gemm_wmma<<<blocks, 256, 0, st>>>(X, ldX, coX, w.p, bias, Y, ldY, coY,
                                      Ncols, w.Npad, w.Kpad, dir);
}

// workspace layout (floats). Requires ws_size >= ~200 MB.
#define F_2   ((size_t)BL * 128)      // 2,097,152
#define F_4   ((size_t)BL * 256)
#define F_XZ  ((size_t)BL * 512)
#define F_T   ((size_t)BL * 680)
#define F_G   ((size_t)BL * 340)
#define F_DBL ((size_t)KDIR * BL * 40)
#define F_DEL ((size_t)KDIR * BL * 256)
#define PW_FLOATS ((size_t)524288)    // 2 MB region -> 1M bf16 capacity for packed weights

extern "C" void kernel_launch(void* const* d_in, const int* in_sizes, int n_in,
                              void* d_out, int out_size, void* d_ws, size_t ws_size,
                              hipStream_t stream)
{
    (void)in_sizes; (void)n_in; (void)out_size; (void)ws_size;
    const float* low    = (const float*)d_in[0];
    const float* high   = (const float*)d_in[1];
    const float* ln_w   = (const float*)d_in[2];
    const float* ln_b   = (const float*)d_in[3];
    const float* temper = (const float*)d_in[4];
    const float* q_c_w  = (const float*)d_in[5];
    const float* q_dw_c = (const float*)d_in[6];
    const float* kv_c_w = (const float*)d_in[7];
    const float* kv_dw_c= (const float*)d_in[8];
    const float* q_t_w  = (const float*)d_in[9];
    const float* q_dw_t = (const float*)d_in[10];
    const float* kv_t_w = (const float*)d_in[11];
    const float* kv_dw_t= (const float*)d_in[12];
    const float* po_c_w = (const float*)d_in[13];
    const float* po_t_w = (const float*)d_in[14];
    const float* cat_w  = (const float*)d_in[15];
    const float* cat_b  = (const float*)d_in[16];
    const float* pin_w  = (const float*)d_in[17];
    const float* ffn_dw = (const float*)d_in[18];
    const float* pout_w = (const float*)d_in[19];
    const float* inpj_w = (const float*)d_in[20];
    const float* c2d_w  = (const float*)d_in[21];
    const float* c2d_b  = (const float*)d_in[22];
    const float* xpj_w  = (const float*)d_in[23];
    const float* dtp_w  = (const float*)d_in[24];
    const float* dtp_b  = (const float*)d_in[25];
    const float* A_logs = (const float*)d_in[26];
    const float* Ds     = (const float*)d_in[27];
    const float* onrm_w = (const float*)d_in[28];
    const float* onrm_b = (const float*)d_in[29];
    const float* opj_w  = (const float*)d_in[30];
    const float* ln1_w  = (const float*)d_in[31];
    const float* ln1_b  = (const float*)d_in[32];
    const float* ln2_w  = (const float*)d_in[33];
    const float* ln2_b  = (const float*)d_in[34];
    const float* skip1  = (const float*)d_in[35];
    const float* skip2  = (const float*)d_in[36];
    const float* cab1_w = (const float*)d_in[37];
    const float* cab1_b = (const float*)d_in[38];
    const float* cab2_w = (const float*)d_in[39];
    const float* cab2_b = (const float*)d_in[40];
    float* out = (float*)d_out;

    float* wsf = (float*)d_ws;
    float* X1  = wsf;                         // residual after attention (later y3)
    float* X2  = wsf + F_2;                   // residual after FFN (= xn)
    bf16_t* pwbase = (bf16_t*)(wsf + 2 * F_2);// packed-weight region
    float* AR  = wsf + 2 * F_2 + PW_FLOATS;   // reusable arena

    // ---- prepack all GEMM weights to bf16 (tiny; recomputed every call, deterministic) ----
    size_t pwoff = 0;
    auto mkpack = [&](const float* Wt, int N, int K) -> PW {
        PW r;
        r.Npad = (N + 15) & ~15;
        r.Kpad = (K + 31) & ~31;
        r.p = pwbase + pwoff;
        size_t elems = (size_t)r.Npad * r.Kpad;
        pwoff = (pwoff + elems + 15) & ~(size_t)15;   // keep 32B alignment
        k_pack<<<eblocks((long long)r.Npad * r.Kpad), 256, 0, stream>>>(Wt, r.p, N, K, r.Npad, r.Kpad);
        return r;
    };
    PW Pqc  = mkpack(q_c_w,  128, 128);
    PW Pkvc = mkpack(kv_c_w, 256, 128);
    PW Pqt  = mkpack(q_t_w,  128, 128);
    PW Pkvt = mkpack(kv_t_w, 256, 128);
    PW Ppoc = mkpack(po_c_w, 128, 128);
    PW Ppot = mkpack(po_t_w, 128, 128);
    PW Pcat = mkpack(cat_w,  128, 256);
    PW Ppin = mkpack(pin_w,  680, 128);
    PW Ppout= mkpack(pout_w, 128, 340);
    PW Pinp = mkpack(inpj_w, 512, 128);
    PW Pxpj[KDIR], Pdtp[KDIR];
    for (int k = 0; k < KDIR; ++k) {
        Pxpj[k] = mkpack(xpj_w + (size_t)k * 40 * 256, 40, 256);
        Pdtp[k] = mkpack(dtp_w + (size_t)k * 256 * 8, 256, 8);
    }
    PW Popj = mkpack(opj_w, 128, 256);

    const int rowsLN = BL;
    const int lnBlocks = (rowsLN + 7) / 8;

    // ========== Phase A: attention ==========
    float* SA = AR;                 // low nlc
    float* SB = AR + F_2;           // high nlc
    float* SC = AR + 2 * F_2;       // ln(low)
    float* SD = AR + 3 * F_2;       // q_c / attn_out
    float* SE = AR + 4 * F_2;       // q_c dw
    float* SH = AR + 5 * F_2;       // q_t
    float* SI = AR + 6 * F_2;       // q_t dw
    float* SL = AR + 7 * F_2;       // out_c
    float* SM = AR + 8 * F_2;       // out_t
    float* SF = AR + 9 * F_2;       // kv_c        (F_4)
    float* SG = SF + F_4;           // kv_c dw     (F_4)
    float* SJ = SG + F_4;           // kv_t        (F_4)
    float* SK = SJ + F_4;           // kv_t dw     (F_4)
    float* SN = SK + F_4;           // [out_c|out_t] (F_4)

    k_nchw2nlc<<<eblocks((long long)BB * CCH * LLEN), 256, 0, stream>>>(low, SA);
    k_nchw2nlc<<<eblocks((long long)BB * CCH * LLEN), 256, 0, stream>>>(high, SB);
    ln_lastdim<<<lnBlocks, 256, 0, stream>>>(SA, ln_w, ln_b, SC, CCH, rowsLN);

    launch_gemm(stream, SB, 128, 0, Pqc,  nullptr, SD, 128, 0, 128, -1);
    dwconv3<<<eblocks((long long)BL * 128), 256, 0, stream>>>(SD, 128, 0, q_dw_c,  nullptr, SE, 128, 0);
    launch_gemm(stream, SB, 128, 0, Pkvc, nullptr, SF, 256, 0, 256, -1);
    dwconv3<<<eblocks((long long)BL * 256), 256, 0, stream>>>(SF, 256, 0, kv_dw_c, nullptr, SG, 256, 0);
    launch_gemm(stream, SC, 128, 0, Pqt,  nullptr, SH, 128, 0, 128, -1);
    dwconv3<<<eblocks((long long)BL * 128), 256, 0, stream>>>(SH, 128, 0, q_dw_t,  nullptr, SI, 128, 0);
    launch_gemm(stream, SC, 128, 0, Pkvt, nullptr, SJ, 256, 0, 256, -1);
    dwconv3<<<eblocks((long long)BL * 256), 256, 0, stream>>>(SJ, 256, 0, kv_dw_t, nullptr, SK, 256, 0);

    attn_half<<<BB * NHEAD, 256, 0, stream>>>(SE, SK, temper, SL);   // out_c = attn(q_c, kv_t)
    attn_half<<<BB * NHEAD, 256, 0, stream>>>(SI, SG, temper, SM);   // out_t = attn(q_t, kv_c)

    launch_gemm(stream, SL, 128, 0, Ppoc, nullptr, SN, 256, 0,   128, -1);
    launch_gemm(stream, SM, 128, 0, Ppot, nullptr, SN, 256, 128, 128, -1);
    launch_gemm(stream, SN, 256, 0, Pcat, cat_b,   SD, 128, 0,   128, -1);
    k_add<<<eblocks((long long)F_2), 256, 0, stream>>>(SA, SD, X1, (long long)F_2);

    // ========== Phase B: FFN ==========
    float* BT  = AR;                    // pin out  [BL,680]
    float* BT2 = AR + F_T;              // dw out   [BL,680]
    float* BG  = BT2 + F_T;             // gate     [BL,340]
    float* BLN = BG + F_G;              // ln(x1)
    float* BF  = BLN + F_2;             // pout out

    ln_lastdim<<<lnBlocks, 256, 0, stream>>>(X1, ln_w, ln_b, BLN, CCH, rowsLN);
    launch_gemm(stream, BLN, 128, 0, Ppin, nullptr, BT, 680, 0, 680, -1);
    dwconv3<<<eblocks((long long)BL * 680), 256, 0, stream>>>(BT, 680, 0, ffn_dw, nullptr, BT2, 680, 0);
    k_gate<<<eblocks((long long)BL * HID), 256, 0, stream>>>(BT2, BG);
    launch_gemm(stream, BG, 340, 0, Ppout, nullptr, BF, 128, 0, 128, -1);
    k_add<<<eblocks((long long)F_2), 256, 0, stream>>>(X1, BF, X2, (long long)F_2);

    // ========== Phase C: SS2D ==========
    float* CLN  = AR;                   // ln(x2)
    float* CXZ  = AR + F_2;             // in_proj out [BL,512]
    float* CXC  = CXZ + F_XZ;           // silu(dwconv(x)) [BL,256]
    float* CDBL = CXC + F_4;            // x_dbl [K][BL][40]
    float* CDEL = CDBL + F_DBL;         // delta [K][BL][256]
    float* CYC  = CDEL + F_DEL;         // combined scan y [BL,256]
    float* CYLN = CYC + F_4;            // ln + gate
    float* CSS  = CYLN + F_4;           // out_proj out

    ln_lastdim<<<lnBlocks, 256, 0, stream>>>(X2, ln1_w, ln1_b, CLN, CCH, rowsLN);
    launch_gemm(stream, CLN, 128, 0, Pinp, nullptr, CXZ, 512, 0, 512, -1);
    dwconv3<<<eblocks((long long)BL * 256), 256, 0, stream>>>(CXZ, 512, 0, c2d_w, c2d_b, CXC, 256, 1);

    for (int k = 0; k < KDIR; ++k) {
        launch_gemm(stream, CXC, 256, 0, Pxpj[k], nullptr,
                    CDBL + (size_t)k * BL * 40, 40, 0, 40, k);       // gathered rows
        launch_gemm(stream, CDBL + (size_t)k * BL * 40, 40, 0, Pdtp[k], nullptr,
                    CDEL + (size_t)k * BL * 256, 256, 0, 256, -1);
    }
    k_softplus<<<eblocks((long long)KDIR * BL * 256), 256, 0, stream>>>(CDEL, dtp_b);
    k_zero<<<eblocks((long long)F_4), 256, 0, stream>>>(CYC, (long long)F_4);
    sscan<<<BB * KDIR, 256, 0, stream>>>(CXC, CDBL, CDEL, A_logs, Ds, CYC);

    ln_lastdim<<<lnBlocks, 256, 0, stream>>>(CYC, onrm_w, onrm_b, CYLN, DINNER, rowsLN);
    k_mulsilu<<<eblocks((long long)BL * 256), 256, 0, stream>>>(CYLN, CXZ);
    launch_gemm(stream, CYLN, 256, 0, Popj, nullptr, CSS, 128, 0, 128, -1);
    k_scaleadd<<<eblocks((long long)F_2), 256, 0, stream>>>(X2, skip1, CSS, X1, (long long)F_2);  // y3 -> X1

    // ========== Phase D: CAB + final ==========
    float* DLN = AR;                    // ln(y3)
    float* DC1 = AR + F_2;              // cab1 out [BL,42]
    float* DC2 = DC1 + (size_t)BL * 42; // cab2 out [BL,128]

    ln_lastdim<<<lnBlocks, 256, 0, stream>>>(X1, ln2_w, ln2_b, DLN, CCH, rowsLN);
    conv3x3<<<eblocks((long long)BL * 42), 256, 0, stream>>>(DLN, 128, cab1_w, cab1_b, DC1, 42, 1);
    conv3x3<<<eblocks((long long)BL * 128), 256, 0, stream>>>(DC1, 42, cab2_w, cab2_b, DC2, 128, 0);
    k_final<<<eblocks((long long)BB * CCH * LLEN), 256, 0, stream>>>(X1, DC2, skip2, out);
}
